// QuantESPCN_558345748723
// MI455X (gfx1250) — 8-bit matrix operand order repaired and checked by reading, compiled, not run on hardware
//
#include <hip/hip_runtime.h>
#include <stdint.h>
#include <stddef.h>

typedef __attribute__((ext_vector_type(8))) int v8i;
typedef __attribute__((ext_vector_type(4))) unsigned int u32x4;

__device__ __forceinline__ float clampf(float x, float lo, float hi) {
  return fminf(fmaxf(x, lo), hi);
}

__global__ void k_quant_input_nhwc(const float* __restrict__ x,
                                   const float* __restrict__ s_in,
                                   uint8_t* __restrict__ q, int n) {
  int i = blockIdx.x * blockDim.x + threadIdx.x;
  if (i < n) {
    const int c = i % 3;
    int t = i / 3;
    const int xx = t % 256; t /= 256;
    const int y = t % 256;
    const int b = t / 256;
    const float v = rintf(x[(((size_t)b * 3 + c) * 256 + y) * 256 + xx] / s_in[0]);
    q[i] = (uint8_t)clampf(v, 0.f, 255.f);
  }
}

template <int FAN, int KPAD>
__global__ void k_quant_w_perchan(const float* __restrict__ w, float nlev,
                                  int8_t* __restrict__ codes,
                                  float* __restrict__ s_w) {
  __shared__ float red[256];
  const int oc = blockIdx.x, tid = threadIdx.x;
  float amax = 0.f;
  for (int i = tid; i < FAN; i += 256) amax = fmaxf(amax, fabsf(w[oc * FAN + i]));
  red[tid] = amax;
  __syncthreads();
  for (int s = 128; s > 0; s >>= 1) {
    if (tid < s) red[tid] = fmaxf(red[tid], red[tid + s]);
    __syncthreads();
  }
  const float s = fmaxf(red[0] / nlev, 1e-8f);
  if (tid == 0) s_w[oc] = s;
  for (int i = tid; i < KPAD; i += 256) {
    float q = 0.f;
    if (i < FAN) q = clampf(rintf(w[oc * FAN + i] / s), -nlev, nlev);
    codes[oc * KPAD + i] = (int8_t)(int)q;
  }
}

template <int CIN, int COUT>
__global__ void k_quant_w_tap(const float* __restrict__ w, float nlev,
                              int8_t* __restrict__ codes,
                              float* __restrict__ s_w) {
  __shared__ float red[256];
  constexpr int FAN = CIN * 9;
  const int oc = blockIdx.x, tid = threadIdx.x;
  float amax = 0.f;
  for (int i = tid; i < FAN; i += 256) amax = fmaxf(amax, fabsf(w[oc * FAN + i]));
  red[tid] = amax;
  __syncthreads();
  for (int s = 128; s > 0; s >>= 1) {
    if (tid < s) red[tid] = fmaxf(red[tid], red[tid + s]);
    __syncthreads();
  }
  const float s = fmaxf(red[0] / nlev, 1e-8f);
  if (tid == 0) s_w[oc] = s;
  for (int i = tid; i < FAN; i += 256) {
    const int ic = i / 9;
    const int kh = (i % 9) / 3;
    const int kw = i % 3;
    const float q = clampf(rintf(w[oc * FAN + i] / s), -nlev, nlev);
    codes[((kh * 3 + kw) * COUT + oc) * CIN + ic] = (int8_t)(int)q;
  }
}

__global__ void k_quant_w4_pair(const float* __restrict__ w,
                                int8_t* __restrict__ codes,
                                float* __restrict__ s_w) {
  __shared__ float red[256];
  const int tid = threadIdx.x;
  float amax = 0.f;
  for (int i = tid; i < 27 * 288; i += 256) amax = fmaxf(amax, fabsf(w[i]));
  red[tid] = amax;
  __syncthreads();
  for (int s = 128; s > 0; s >>= 1) {
    if (tid < s) red[tid] = fmaxf(red[tid], red[tid + s]);
    __syncthreads();
  }
  const float s = fmaxf(red[0] / 127.f, 1e-8f);
  if (tid == 0) s_w[0] = s;
  for (int i = tid; i < 6 * 32 * 64; i += 256) {
    const int k = i & 63;
    const int oc = (i >> 6) & 31;
    const int step = i >> 11;
    const int kh = step >> 1, p = step & 1;
    const int ch = k & 31, kwoff = k >> 5;
    const int kw = 2 * p + kwoff;
    float q = 0.f;
    if (oc < 27 && kw < 3)
      q = clampf(rintf(w[((oc * 32 + ch) * 3 + kh) * 3 + kw] / s), -127.f, 127.f);
    codes[i] = (int8_t)(int)q;
  }
}

__global__ void k_prep_affine(const float* __restrict__ s_w,
                              const float* __restrict__ bias,
                              const float* __restrict__ g,
                              const float* __restrict__ be,
                              const float* __restrict__ m,
                              const float* __restrict__ v,
                              const float* __restrict__ s_act,
                              const float* __restrict__ s_next,
                              float* __restrict__ A, float* __restrict__ B,
                              int cout, int per_tensor, int has_bn) {
  const int oc = threadIdx.x;
  if (oc >= cout) return;
  const float sw = per_tensor ? s_w[0] : s_w[oc];
  const float sa = s_act[0], sn = s_next[0];
  if (has_bn) {
    const float inv = g[oc] * rsqrtf(v[oc] + 1e-5f);
    A[oc] = sa * sw * inv / sn;
    B[oc] = (bias[oc] * inv + be[oc] - m[oc] * inv) / sn;
  } else {
    A[oc] = sa * sw / sn;
    B[oc] = 0.f;
  }
}

__global__ __launch_bounds__(256) void k_conv1(
    const uint8_t* __restrict__ act_in, const int8_t* __restrict__ wcodes,
    const float* __restrict__ Asc, const float* __restrict__ Bbs,
    uint8_t* __restrict__ act_out) {
  constexpr int CIN = 3, COUT = 64, KSZ = 5, PAD = 2, K = 75, KPAD = 128;
  constexpr int PW = 16 + KSZ - 1;
  constexpr int PATCH = KSZ * PW * CIN;
  constexpr int NT = COUT / 16;

  __shared__ __align__(16) int8_t s_wts[COUT * KPAD];
  __shared__ __align__(16) uint8_t s_patch[8][PATCH];

  const int tid = threadIdx.x;
  const int wave = tid >> 5, lane = tid & 31;
  const int hl = lane >> 4, mcol = lane & 15;
  const int bidx = blockIdx.x;
  const int b = bidx >> 9;
  const int row = (bidx >> 1) & 255;
  const int ox0 = (((bidx & 1) << 3) + wave) << 4;

  {
    const int* src = (const int*)wcodes;
    int* dst = (int*)s_wts;
    for (int i = tid; i < (COUT * KPAD) / 4; i += 256) dst[i] = src[i];
  }
  {
    uint8_t* p = s_patch[wave];
    for (int i = lane; i < PATCH; i += 32) {
      const int ic = i % 3;
      const int pix = i / 3;
      const int r = pix / PW, col = pix % PW;
      const int iy = row + r - PAD, ix = ox0 + col - PAD;
      uint8_t val = 0;
      if (iy >= 0 && iy < 256 && ix >= 0 && ix < 256)
        val = act_in[(((size_t)b * 256 + iy) * 256 + ix) * 3 + ic];
      p[i] = val;
    }
  }
  __syncthreads();

  v8i acc[NT];
#pragma unroll
  for (int nt = 0; nt < NT; ++nt) { v8i z = {0,0,0,0,0,0,0,0}; acc[nt] = z; }

  const uint8_t* patch = s_patch[wave];
#pragma unroll
  for (int ks = 0; ks < KPAD / 64; ++ks) {
    v8i a;
#pragma unroll
    for (int vq = 0; vq < 8; ++vq) {
      const int kb = ks*64 + 32*(vq>>2) + 16*hl + 4*(vq&3);
      uint32_t d = 0;
#pragma unroll
      for (int by = 0; by < 4; ++by) {
        const int k = kb + by;
        uint32_t q = 0;
        if (k < K) {
          const int ic = k / 25, rr = k % 25;
          const int kh = rr / 5, kw = rr % 5;
          q = patch[((kh * PW) + mcol + kw) * 3 + ic];
        }
        d |= q << (8 * by);
      }
      a[vq] = (int)d;
    }
#pragma unroll
    for (int nt = 0; nt < NT; ++nt) {
      v8i bm;
#pragma unroll
      for (int vq = 0; vq < 8; ++vq) {
        const int koff = ks*64 + 32*(vq>>2) + 16*hl + 4*(vq&3);
        bm[vq] = *(const int*)&s_wts[(nt * 16 + mcol) * KPAD + koff];
      }
      acc[nt] = __builtin_amdgcn_wmma_i32_16x16x64_iu8(false, a, true, bm,
                                                       acc[nt], false, false);
    }
  }

#pragma unroll
  for (int nt = 0; nt < NT; ++nt) {
    const int oc = nt * 16 + mcol;
    const float Aoc = Asc[oc], Boc = Bbs[oc];
#pragma unroll
    for (int r = 0; r < 8; ++r) {
      const float q = clampf(rintf((float)acc[nt][r] * Aoc + Boc), 0.f, 15.f);
      const int pix = ox0 + r + 8 * hl;
      act_out[(((size_t)b * 256 + row) * 256 + pix) * COUT + oc] = (uint8_t)q;
    }
  }
}

template <int CIN, int COUT, int COUT_REAL, int STEPS, bool PAIRED, int QMAX,
          bool SHUFFLE>
__global__ __launch_bounds__(256) void k_conv_tap(
    const uint8_t* __restrict__ act_in, const int8_t* __restrict__ wcodes,
    const float* __restrict__ Asc, const float* __restrict__ Bbs,
    uint8_t* __restrict__ act_out, float* __restrict__ out_f,
    const float* __restrict__ s_out_p) {
  constexpr int PAD = 1;
  constexpr int PW = PAIRED ? 19 : 18;
  constexpr int NPIX = 3 * PW;
  constexpr int NT = COUT / 16;
  constexpr int VPP = CIN / 16;

  __shared__ __align__(16) int8_t s_wts[STEPS * COUT * 64];
  __shared__ __align__(16) uint8_t s_patch[8][NPIX * CIN];

  const int tid = threadIdx.x;
  const int wave = tid >> 5, lane = tid & 31;
  const int hl = lane >> 4, mcol = lane & 15;
  const int bidx = blockIdx.x;
  const int b = bidx >> 9;
  const int row = (bidx >> 1) & 255;
  const int ox0 = (((bidx & 1) << 3) + wave) << 4;

  {
    const int* src = (const int*)wcodes;
    int* dst = (int*)s_wts;
    for (int i = tid; i < (STEPS * COUT * 64) / 4; i += 256) dst[i] = src[i];
  }
  {
    uint8_t* p = s_patch[wave];
    for (int j = lane; j < NPIX * VPP; j += 32) {
      const int pix = j / VPP, vs = j % VPP;
      const int r = pix / PW, col = pix % PW;
      const int iy = row + r - PAD, ix = ox0 + col - PAD;
      u32x4 val = {0u, 0u, 0u, 0u};
      if (iy >= 0 && iy < 256 && ix >= 0 && ix < 256)
        val = *(const u32x4*)&act_in[(((size_t)b * 256 + iy) * 256 + ix) * CIN +
                                     vs * 16];
      *(u32x4*)&p[pix * CIN + vs * 16] = val;
    }
  }
  __syncthreads();

  v8i acc[NT];
#pragma unroll
  for (int nt = 0; nt < NT; ++nt) { v8i z = {0,0,0,0,0,0,0,0}; acc[nt] = z; }

  const uint8_t* patch = s_patch[wave];
#pragma unroll
  for (int step = 0; step < STEPS; ++step) {
    v8i a;
    if constexpr (!PAIRED) {
      const int kh = step / 3, kw = step % 3;
      const int pbase = ((kh * PW) + mcol + kw) * CIN;
#pragma unroll
      for (int vq = 0; vq < 8; ++vq) {
        const int c = 32*(vq>>2) + 16*hl + 4*(vq&3);
        a[vq] = *(const int*)&patch[pbase + c];
      }
    } else {
      const int kh = step / 2, p = step % 2;
#pragma unroll
      for (int vq = 0; vq < 8; ++vq) {
        const int c = 32*(vq>>2) + 16*hl + 4*(vq&3);
        const int kw = 2 * p + (c >> 5);
        const int ch = c & 31;
        a[vq] = *(const int*)&patch[((kh * PW) + mcol + kw) * CIN + ch];
      }
    }
#pragma unroll
    for (int nt = 0; nt < NT; ++nt) {
      v8i bm;
#pragma unroll
      for (int vq = 0; vq < 8; ++vq) {
        const int koff = 32*(vq>>2) + 16*hl + 4*(vq&3);
        bm[vq] =
            *(const int*)&s_wts[((step * COUT) + nt * 16 + mcol) * 64 + koff];
      }
      acc[nt] = __builtin_amdgcn_wmma_i32_16x16x64_iu8(false, a, true, bm,
                                                       acc[nt], false, false);
    }
  }

#pragma unroll
  for (int nt = 0; nt < NT; ++nt) {
    const int oc = nt * 16 + mcol;
    const float Aoc = Asc[oc], Boc = Bbs[oc];
    if constexpr (!SHUFFLE) {
#pragma unroll
      for (int r = 0; r < 8; ++r) {
        const float q =
            clampf(rintf((float)acc[nt][r] * Aoc + Boc), 0.f, (float)QMAX);
        const int pix = ox0 + r + 8 * hl;
        act_out[(((size_t)b * 256 + row) * 256 + pix) * COUT_REAL + oc] =
            (uint8_t)q;
      }
    } else {
      if (oc < COUT_REAL) {
        const float so = s_out_p[0];
        const int c2 = oc / 9, rem9 = oc % 9;
        const int ry = rem9 / 3, rx = rem9 % 3;
#pragma unroll
        for (int r = 0; r < 8; ++r) {
          const int m = r + 8 * hl;
          const float q =
              clampf(rintf((float)acc[nt][r] * Aoc + Boc), 0.f, 255.f);
          const size_t o = (((size_t)b * 3 + c2) * 768 + (row * 3 + ry)) * 768 +
                           (size_t)(ox0 + m) * 3 + rx;
          out_f[o] = q * so;
        }
      }
    }
  }
}

extern "C" void kernel_launch(void* const* d_in, const int* in_sizes, int n_in,
                              void* d_out, int out_size, void* d_ws,
                              size_t ws_size, hipStream_t stream) {
  (void)in_sizes; (void)n_in; (void)out_size; (void)ws_size;
  const float* x  = (const float*)d_in[0];
  const float* w1 = (const float*)d_in[1];
  const float* b1 = (const float*)d_in[2];
  const float* w2 = (const float*)d_in[3];
  const float* b2 = (const float*)d_in[4];
  const float* w3 = (const float*)d_in[5];
  const float* b3 = (const float*)d_in[6];
  const float* w4 = (const float*)d_in[7];
  const float* g1 = (const float*)d_in[8];
  const float* be1 = (const float*)d_in[9];
  const float* m1 = (const float*)d_in[10];
  const float* v1 = (const float*)d_in[11];
  const float* g2 = (const float*)d_in[12];
  const float* be2 = (const float*)d_in[13];
  const float* m2 = (const float*)d_in[14];
  const float* v2 = (const float*)d_in[15];
  const float* g3 = (const float*)d_in[16];
  const float* be3 = (const float*)d_in[17];
  const float* m3 = (const float*)d_in[18];
  const float* v3 = (const float*)d_in[19];
  const float* s_in = (const float*)d_in[20];
  const float* s2 = (const float*)d_in[21];
  const float* s3 = (const float*)d_in[22];
  const float* s4 = (const float*)d_in[23];
  const float* s_out = (const float*)d_in[24];

  char* ws = (char*)d_ws;
  size_t off = 0;
  auto carve = [&](size_t bytes) {
    char* p = ws + off;
    off += (bytes + 255) & ~(size_t)255;
    return p;
  };
  uint8_t* qx   = (uint8_t*)carve((size_t)8 * 256 * 256 * 3);
  uint8_t* act1 = (uint8_t*)carve((size_t)8 * 256 * 256 * 64);
  uint8_t* act2 = (uint8_t*)carve((size_t)8 * 256 * 256 * 64);
  uint8_t* act3 = (uint8_t*)carve((size_t)8 * 256 * 256 * 32);
  int8_t* wc1 = (int8_t*)carve(64 * 128);
  int8_t* wc2 = (int8_t*)carve(9 * 64 * 64);
  int8_t* wc3 = (int8_t*)carve(9 * 32 * 64);
  int8_t* wc4 = (int8_t*)carve(6 * 32 * 64);
  float* sw1 = (float*)carve(64 * 4);
  float* sw2 = (float*)carve(64 * 4);
  float* sw3 = (float*)carve(32 * 4);
  float* sw4 = (float*)carve(4);
  float* A1 = (float*)carve(64 * 4); float* B1 = (float*)carve(64 * 4);
  float* A2 = (float*)carve(64 * 4); float* B2 = (float*)carve(64 * 4);
  float* A3 = (float*)carve(64 * 4); float* B3 = (float*)carve(64 * 4);
  float* A4 = (float*)carve(64 * 4); float* B4 = (float*)carve(64 * 4);

  {
    const int n = 8 * 256 * 256 * 3;
    k_quant_input_nhwc<<<(n + 255) / 256, 256, 0, stream>>>(x, s_in, qx, n);
  }
  k_quant_w_perchan<75, 128><<<64, 256, 0, stream>>>(w1, 127.f, wc1, sw1);
  k_quant_w_tap<64, 64><<<64, 256, 0, stream>>>(w2, 7.f, wc2, sw2);
  k_quant_w_tap<64, 32><<<32, 256, 0, stream>>>(w3, 7.f, wc3, sw3);
  k_quant_w4_pair<<<1, 256, 0, stream>>>(w4, wc4, sw4);
  k_prep_affine<<<1, 64, 0, stream>>>(sw1, b1, g1, be1, m1, v1, s_in, s2, A1, B1, 64, 0, 1);
  k_prep_affine<<<1, 64, 0, stream>>>(sw2, b2, g2, be2, m2, v2, s2, s3, A2, B2, 64, 0, 1);
  k_prep_affine<<<1, 64, 0, stream>>>(sw3, b3, g3, be3, m3, v3, s3, s4, A3, B3, 32, 0, 1);
  k_prep_affine<<<1, 64, 0, stream>>>(sw4, nullptr, nullptr, nullptr, nullptr,
                                      nullptr, s4, s_out, A4, B4, 32, 1, 0);
  k_conv1<<<4096, 256, 0, stream>>>(qx, wc1, A1, B1, act1);
  k_conv_tap<64, 64, 64, 9, false, 15, false>
      <<<4096, 256, 0, stream>>>(act1, wc2, A2, B2, act2, nullptr, nullptr);
  k_conv_tap<64, 32, 32, 9, false, 15, false>
      <<<4096, 256, 0, stream>>>(act2, wc3, A3, B3, act3, nullptr, nullptr);
  k_conv_tap<32, 32, 27, 6, true, 255, true>
      <<<4096, 256, 0, stream>>>(act3, wc4, A4, B4, nullptr, (float*)d_out,
                                 s_out);
}
